// HypLinear_16131897164125
// MI455X (gfx1250) — matrix operand order repaired and checked by reading, not compiled, not run on hardware
//
#include <hip/hip_runtime.h>
#include <cstdint>
#include <cstddef>

// ---------------------------------------------------------------------------
// Types for CDNA5 WMMA / TDM builtins
// ---------------------------------------------------------------------------
typedef __attribute__((ext_vector_type(16))) __bf16       v16bf;
typedef __attribute__((ext_vector_type(8)))  float        v8f;
typedef __attribute__((ext_vector_type(4)))  unsigned int v4u;
typedef __attribute__((ext_vector_type(8)))  int          v8i;
typedef __attribute__((ext_vector_type(4)))  int          v4i;

// Problem sizes (fixed by the reference)
#define DIM_M 8192
#define DIM_N 4096
#define DIM_K 4096

// GEMM tiling
#define BM 128
#define BN 64
#define BK 32
#define LDSW 36          // padded LDS row stride in dwords (32 data + 4 pad) -> 16B alignment kept
#define BUF_FLTS ((BM + BN) * LDSW)   // floats per ping-pong buffer (6912)

#define EPS_BND   1e-5f
#define MIN_NORM  1e-15f
#define MAXNORM   (1.0f - EPS_BND)

// ---------------------------------------------------------------------------
// TDM: 2D tile load Global -> LDS via Tensor DMA Descriptor (D#).
// data_size = 4B; pad: +4 dwords after every 32 dwords stored -> LDS row stride 36.
// ---------------------------------------------------------------------------
__device__ __forceinline__ void tdm_load_tile_2d(unsigned lds_off,
                                                 const float* gaddr,
                                                 unsigned tensor_d0,   // fast dim (elements)
                                                 unsigned tensor_d1,   // slow dim (elements)
                                                 unsigned tile_d0,     // fast tile dim
                                                 unsigned tile_d1,     // slow tile dim
                                                 unsigned stride0) {   // elements between rows
    unsigned long long ga = (unsigned long long)(size_t)gaddr;
    v4u g0;
    g0[0] = 1u;                                               // count = 1 (valid descriptor)
    g0[1] = lds_off;                                          // LDS byte address
    g0[2] = (unsigned)(ga & 0xFFFFFFFFu);                     // global addr [31:0]
    g0[3] = (unsigned)((ga >> 32) & 0x01FFFFFFu)              // global addr [56:32]
          | 0x80000000u;                                      // type = 2 at bits [127:126]
    v8i g1;
    g1[0] = (int)((2u << 16)            // data_size = 4 bytes
                | (1u << 20)            // pad_enable
                | (4u << 22)            // pad_interval: 32 dwords
                | (3u << 25));          // pad_amount: 4 dwords
    g1[1] = (int)((tensor_d0 & 0xFFFFu) << 16);
    g1[2] = (int)(((tensor_d1 & 0xFFFFu) << 16) | ((tensor_d0 >> 16) & 0xFFFFu));
    g1[3] = (int)(((tile_d0  & 0xFFFFu) << 16) | ((tensor_d1 >> 16) & 0xFFFFu));
    g1[4] = (int)(tile_d1 & 0xFFFFu);
    g1[5] = (int)stride0;
    g1[6] = 0;
    g1[7] = 0;
    v4i z4 = {0, 0, 0, 0};
#if defined(__clang_major__) && (__clang_major__ >= 23)
    v8i z8 = {0, 0, 0, 0, 0, 0, 0, 0};
    __builtin_amdgcn_tensor_load_to_lds(g0, g1, z4, z4, z8, 0);
#else
    __builtin_amdgcn_tensor_load_to_lds(g0, g1, z4, z4, 0);
#endif
}

__device__ __forceinline__ float artanhf_(float z) {
    z = fminf(fmaxf(z, -1.0f + 1e-7f), 1.0f - 1e-7f);
    return 0.5f * logf((1.0f + z) / (1.0f - z));
}

// split 8 fp32 (two float4) into bf16 hi/lo halves of a v16bf fragment
__device__ __forceinline__ void bf_split8(float4 a, float4 b, v16bf& H, v16bf& L, int base) {
    float f[8] = { a.x, a.y, a.z, a.w, b.x, b.y, b.z, b.w };
#pragma unroll
    for (int j = 0; j < 8; ++j) {
        __bf16 h = (__bf16)f[j];
        H[base + j] = h;
        L[base + j] = (__bf16)(f[j] - (float)h);
    }
}

// ---------------------------------------------------------------------------
// Kernel 1: bias -> bman = expmap0(bias), y2 = ||bman||^2
// ---------------------------------------------------------------------------
__global__ __launch_bounds__(256) void prep_bias_kernel(const float* __restrict__ bias,
                                                        float* __restrict__ bman,
                                                        float* __restrict__ y2out) {
    __shared__ float sdata[256];
    const int tid = threadIdx.x;
    float s = 0.0f;
    for (int j = tid; j < DIM_N; j += 256) { float b = bias[j]; s += b * b; }
    sdata[tid] = s; __syncthreads();
    for (int st = 128; st > 0; st >>= 1) { if (tid < st) sdata[tid] += sdata[tid + st]; __syncthreads(); }
    const float bn = fmaxf(sqrtf(sdata[0]), MIN_NORM);
    const float scale = tanhf(bn) / bn;                 // sqrt(C)=1
    __syncthreads();
    float s2 = 0.0f;
    for (int j = tid; j < DIM_N; j += 256) { float v = scale * bias[j]; bman[j] = v; s2 += v * v; }
    sdata[tid] = s2; __syncthreads();
    for (int st = 128; st > 0; st >>= 1) { if (tid < st) sdata[tid] += sdata[tid + st]; __syncthreads(); }
    if (tid == 0) y2out[0] = sdata[0];
}

// ---------------------------------------------------------------------------
// Kernel 2: per-row L2 norm of x (raw, clamped) -> xnorm[8192]
// ---------------------------------------------------------------------------
__global__ __launch_bounds__(256) void rownorm_x_kernel(const float* __restrict__ x,
                                                        float* __restrict__ xnorm) {
    __shared__ float sdata[256];
    const int row = blockIdx.x, tid = threadIdx.x;
    const float4* p = (const float4*)(x + (size_t)row * DIM_K);   // 1024 float4 per row
    float s = 0.0f;
#pragma unroll
    for (int t = 0; t < 4; ++t) {
        float4 v = p[tid + t * 256];
        s += v.x * v.x + v.y * v.y + v.z * v.z + v.w * v.w;
    }
    sdata[tid] = s; __syncthreads();
    for (int st = 128; st > 0; st >>= 1) { if (tid < st) sdata[tid] += sdata[tid + st]; __syncthreads(); }
    if (tid == 0) xnorm[row] = fmaxf(sqrtf(sdata[0]), MIN_NORM);
}

// ---------------------------------------------------------------------------
// Kernel 3: GEMM out = x @ W^T.  bf16x3 split on v_wmma_f32_16x16x32_bf16.
// Double-buffered TDM: tile i+1 DMAs into the ping-pong buffer while tile i
// is consumed (s_wait_tensorcnt 2 exploits in-order TENSORcnt completion).
// 256 threads (8 waves); WG tile 128M x 64N; wave tile 32x32 (2x2 frags).
// ---------------------------------------------------------------------------
__global__ __launch_bounds__(256) void gemm_kernel(const float* __restrict__ X,
                                                   const float* __restrict__ W,
                                                   float* __restrict__ Out) {
    __shared__ float smem[2 * BUF_FLTS];                // 54 KB (2 x (A 128x36 + B 64x36))

    const int tid      = threadIdx.x;
    const int lane     = tid & 31;
    const int wid      = __builtin_amdgcn_readfirstlane(tid >> 5);  // uniform -> scalar branch
    const int waveM    = (wid & 3) << 5;     // 0,32,64,96
    const int waveN    = (wid >> 2) << 5;    // 0,32
    const int laneHalf = lane >> 4;
    const int laneIdx  = lane & 15;

    const int blockM = blockIdx.y * BM;
    const int blockN = blockIdx.x * BN;

    const unsigned ldsBase = (unsigned)(size_t)(void*)smem;  // low 32 bits = LDS offset
    const float* Xblk = X + (size_t)blockM * DIM_K;
    const float* Wblk = W + (size_t)blockN * DIM_K;

    v8f acc[2][2];
#pragma unroll
    for (int mi = 0; mi < 2; ++mi)
#pragma unroll
        for (int ni = 0; ni < 2; ++ni)
            acc[mi][ni] = (v8f){0.f,0.f,0.f,0.f,0.f,0.f,0.f,0.f};

    const int T = DIM_K / BK;                 // 128 k-tiles

    // prologue: DMA tile 0 into buffer 0
    if (wid == 0) {
        tdm_load_tile_2d(ldsBase,                    Xblk, DIM_K, DIM_M, BK, BM, DIM_K);
        tdm_load_tile_2d(ldsBase + BM * LDSW * 4,    Wblk, DIM_K, DIM_N, BK, BN, DIM_K);
    }

    for (int i = 0; i < T; ++i) {
        const int cur = i & 1;
        float* sA = smem + cur * BUF_FLTS;
        float* sB = sA + BM * LDSW;

        if (wid == 0) {
            if (i + 1 < T) {
                // DMA next tile into the other buffer (its readers finished at the
                // end-of-iteration barrier of i-1), then wait for current tile only.
                const int nxt = cur ^ 1;
                const unsigned ldsN = ldsBase + (unsigned)nxt * (BUF_FLTS * 4);
                const int k1 = (i + 1) * BK;
                tdm_load_tile_2d(ldsN,                 Xblk + k1, DIM_K, DIM_M, BK, BM, DIM_K);
                tdm_load_tile_2d(ldsN + BM * LDSW * 4, Wblk + k1, DIM_K, DIM_N, BK, BN, DIM_K);
                __builtin_amdgcn_s_wait_tensorcnt(2);
            } else {
                __builtin_amdgcn_s_wait_tensorcnt(0);
            }
        }
        __syncthreads();                                 // current tile visible to all waves

        // ---- A fragments (16x32 bf16: lane h=lane/16 holds K [8h,8h+8) U [16+8h,16+8h+8)) ----
        v16bf aH[2], aL[2];
#pragma unroll
        for (int mi = 0; mi < 2; ++mi) {
            const int row = waveM + mi * 16 + laneIdx;
            const float* p = &sA[row * LDSW + laneHalf * 8];
            bf_split8(*(const float4*)(p),      *(const float4*)(p + 4),  aH[mi], aL[mi], 0);
            bf_split8(*(const float4*)(p + 16), *(const float4*)(p + 20), aH[mi], aL[mi], 8);
        }

        // ---- B fragments (32x16 bf16: lane h=lane/16 holds K [8h,8h+8) U [16+8h,16+8h+8)) ----
#pragma unroll
        for (int ni = 0; ni < 2; ++ni) {
            const int col = waveN + ni * 16 + laneIdx;   // column n == row of W
            const float* p = &sB[col * LDSW + laneHalf * 8];
            v16bf bH, bL;
            bf_split8(*(const float4*)(p),     *(const float4*)(p + 4),  bH, bL, 0);
            bf_split8(*(const float4*)(p + 16), *(const float4*)(p + 20), bH, bL, 8);
#pragma unroll
            for (int mi = 0; mi < 2; ++mi) {
                acc[mi][ni] = __builtin_amdgcn_wmma_f32_16x16x32_bf16(
                    false, aH[mi], false, bH, (short)0, acc[mi][ni], false, false);
                acc[mi][ni] = __builtin_amdgcn_wmma_f32_16x16x32_bf16(
                    false, aH[mi], false, bL, (short)0, acc[mi][ni], false, false);
                acc[mi][ni] = __builtin_amdgcn_wmma_f32_16x16x32_bf16(
                    false, aL[mi], false, bH, (short)0, acc[mi][ni], false, false);
            }
        }
        __syncthreads();                                 // buf[cur] free before iter i+1 re-DMAs it
    }

    // ---- store D (VGPR r -> M=r (+8 for lanes 16-31), N=lane%16); NT: don't pollute L2 ----
#pragma unroll
    for (int mi = 0; mi < 2; ++mi)
#pragma unroll
        for (int ni = 0; ni < 2; ++ni)
#pragma unroll
            for (int r = 0; r < 8; ++r) {
                const int m = blockM + waveM + mi * 16 + laneHalf * 8 + r;
                const int n = blockN + waveN + ni * 16 + laneIdx;
                __builtin_nontemporal_store(acc[mi][ni][r], &Out[(size_t)m * DIM_N + n]);
            }
}

// ---------------------------------------------------------------------------
// Kernel 4: fused per-row finish. One block per row:
//   ms = ||mx||^2 ; g = tanh((|mx|/|x|)*artanh(min(|x|,maxnorm)))/|mx| (0 if mx==0)
//   res = g*mx ; x2 = ||res||^2 ; xy = <res,b>
//   out = t * (a1*res + b1*b)/den   with the projection norm computed analytically.
// ---------------------------------------------------------------------------
__global__ __launch_bounds__(256) void rowfinish_kernel(float* __restrict__ out,
                                                        const float* __restrict__ bman,
                                                        const float* __restrict__ xnorm,
                                                        const float* __restrict__ y2p) {
    __shared__ float s1[256];
    __shared__ float s2[256];
    const int row = blockIdx.x, tid = threadIdx.x;
    float4* prow = (float4*)(out + (size_t)row * DIM_N);      // 1024 float4
    const float4* brow = (const float4*)bman;

    float4 mv[4];
    float ssum = 0.0f;
#pragma unroll
    for (int t = 0; t < 4; ++t) {
        mv[t] = prow[tid + t * 256];
        ssum += mv[t].x * mv[t].x + mv[t].y * mv[t].y + mv[t].z * mv[t].z + mv[t].w * mv[t].w;
    }
    s1[tid] = ssum; __syncthreads();
    for (int st = 128; st > 0; st >>= 1) { if (tid < st) s1[tid] += s1[tid + st]; __syncthreads(); }
    const float ms = s1[0];
    __syncthreads();                              // s1 will be reused

    const float xn  = xnorm[row];                 // >= MIN_NORM (raw norm; projection folded in)
    const float mxn = fmaxf(sqrtf(ms), MIN_NORM);
    float g = tanhf((mxn / xn) * artanhf_(fminf(xn, MAXNORM))) / mxn;
    if (ms == 0.0f) g = 0.0f;

    float4 bv[4];
    float a2 = 0.0f, ab = 0.0f;
#pragma unroll
    for (int t = 0; t < 4; ++t) {
        bv[t] = brow[tid + t * 256];
        mv[t].x *= g; mv[t].y *= g; mv[t].z *= g; mv[t].w *= g;
        a2 += mv[t].x * mv[t].x + mv[t].y * mv[t].y + mv[t].z * mv[t].z + mv[t].w * mv[t].w;
        ab += mv[t].x * bv[t].x + mv[t].y * bv[t].y + mv[t].z * bv[t].z + mv[t].w * bv[t].w;
    }
    s1[tid] = a2; s2[tid] = ab; __syncthreads();
    for (int st = 128; st > 0; st >>= 1) {
        if (tid < st) { s1[tid] += s1[tid + st]; s2[tid] += s2[tid + st]; }
        __syncthreads();
    }
    const float x2 = s1[0], xy = s2[0], y2 = y2p[0];

    const float a1 = 1.0f + 2.0f * xy + y2;       // C = 1
    const float b1 = 1.0f - x2;
    float den = 1.0f + 2.0f * xy + x2 * y2;
    den = fmaxf(den, MIN_NORM);
    const float inv = 1.0f / den;
    const float n2 = (a1 * a1 * x2 + 2.0f * a1 * b1 * xy + b1 * b1 * y2) * inv * inv;
    const float on = fmaxf(sqrtf(fmaxf(n2, 0.0f)), MIN_NORM);
    const float tp = (on > MAXNORM) ? (MAXNORM / on) : 1.0f;
    const float f  = tp * inv;

#pragma unroll
    for (int t = 0; t < 4; ++t) {
        float4 o;
        o.x = f * (a1 * mv[t].x + b1 * bv[t].x);
        o.y = f * (a1 * mv[t].y + b1 * bv[t].y);
        o.z = f * (a1 * mv[t].z + b1 * bv[t].z);
        o.w = f * (a1 * mv[t].w + b1 * bv[t].w);
        prow[tid + t * 256] = o;
    }
}

// ---------------------------------------------------------------------------
// Host-side launcher
// ---------------------------------------------------------------------------
extern "C" void kernel_launch(void* const* d_in, const int* in_sizes, int n_in,
                              void* d_out, int out_size, void* d_ws, size_t ws_size,
                              hipStream_t stream) {
    const float* x    = (const float*)d_in[0];   // [8192, 4096]
    const float* wgt  = (const float*)d_in[1];   // [4096, 4096]
    const float* bias = (const float*)d_in[2];   // [4096]
    float* out = (float*)d_out;                  // [8192, 4096]

    // workspace layout (floats)
    float* ws    = (float*)d_ws;
    float* xnorm = ws;                 // 8192
    float* bman  = ws + 8192;          // 4096
    float* y2p   = ws + 12288;         // 1

    prep_bias_kernel<<<1, 256, 0, stream>>>(bias, bman, y2p);
    rownorm_x_kernel<<<DIM_M, 256, 0, stream>>>(x, xnorm);

    dim3 ggrid(DIM_N / BN, DIM_M / BM);          // 64 x 64
    gemm_kernel<<<ggrid, 256, 0, stream>>>(x, wgt, out);

    rowfinish_kernel<<<DIM_M, 256, 0, stream>>>(out, bman, xnorm, y2p);
}
